// KPCL_7232724926714
// MI455X (gfx1250) — matrix operand order repaired and checked by reading, not compiled, not run on hardware
//
#include <hip/hip_runtime.h>
#include <hip/hip_fp16.h>
#include <math.h>

#define N_ROWS 8192
#define D_IN   512
#define D_PROJ 128
#define INV_TEMP (1.0f / 0.15f)
#define LOG2F_CONST 0.6931471805599453f

typedef _Float16 v16h __attribute__((ext_vector_type(16)));
typedef _Float16 v8h  __attribute__((ext_vector_type(8)));
typedef float    v8f  __attribute__((ext_vector_type(8)));

__device__ __forceinline__ float sign_f(float x) {
    return (x > 0.f) ? 1.f : ((x < 0.f) ? -1.f : 0.f);
}

// Tree reduction over 128 threads using a shared scratch array.
__device__ __forceinline__ float block_reduce_128(float v, float* red, int tid) {
    red[tid] = v;
    __syncthreads();
#pragma unroll
    for (int s = 64; s > 0; s >>= 1) {
        if (tid < s) red[tid] += red[tid + s];
        __syncthreads();
    }
    float r = red[0];
    __syncthreads();
    return r;
}

// -----------------------------------------------------------------------------
// Kernel 1: noise augmentation (x2) + MLP projection + L2 normalize -> Zh (f16)
// One block (128 threads) per row. Streaming phase reads ~48 MB total
// (~2 us at 23.3 TB/s); compute phase is 512+128 FMAs/thread against
// L2-resident W1/W2 (256 KB + 64 KB).
// -----------------------------------------------------------------------------
__global__ __launch_bounds__(128) void augment_project_kernel(
    const float* __restrict__ features, const float* __restrict__ noise1,
    const float* __restrict__ noise2,  const float* __restrict__ W1,
    const float* __restrict__ b1,      const float* __restrict__ W2,
    const float* __restrict__ b2,      _Float16* __restrict__ Zh)
{
    __shared__ float x2s[D_IN];
    __shared__ float hs[D_PROJ];
    __shared__ float red[128];

    const int row = blockIdx.x;
    const int tid = threadIdx.x;

    const float4 fv  = ((const float4*)(features + (size_t)row * D_IN))[tid];
    const float4 n1v = ((const float4*)(noise1   + (size_t)row * D_IN))[tid];
    const float4 n2v = ((const float4*)(noise2   + (size_t)row * D_IN))[tid];

    float s1 = n1v.x*n1v.x + n1v.y*n1v.y + n1v.z*n1v.z + n1v.w*n1v.w;
    float s2 = n2v.x*n2v.x + n2v.y*n2v.y + n2v.z*n2v.z + n2v.w*n2v.w;

    const float nrm1 = block_reduce_128(s1, red, tid);
    const float nrm2 = block_reduce_128(s2, red, tid);
    const float inv1 = 0.1f / fmaxf(sqrtf(nrm1), 1e-8f);
    const float inv2 = 0.1f / fmaxf(sqrtf(nrm2), 1e-8f);

    float x1[4], x2[4];
    const float fa[4]  = {fv.x, fv.y, fv.z, fv.w};
    const float na1[4] = {n1v.x, n1v.y, n1v.z, n1v.w};
    const float na2[4] = {n2v.x, n2v.y, n2v.z, n2v.w};
#pragma unroll
    for (int k = 0; k < 4; ++k) {
        x1[k] = fa[k] + sign_f(fa[k]) * na1[k] * inv1;
        x2[k] = x1[k] + sign_f(x1[k]) * na2[k] * inv2;
        x2s[tid * 4 + k] = x2[k];
    }
    __syncthreads();

    // h = relu(x2 @ W1 + b1); thread tid owns output column tid
    float acc = b1[tid];
#pragma unroll 8
    for (int k = 0; k < D_IN; ++k)
        acc = fmaf(x2s[k], W1[(size_t)k * D_PROJ + tid], acc);
    acc = fmaxf(acc, 0.f);
    hs[tid] = acc;
    __syncthreads();

    // p = h @ W2 + b2
    float p = b2[tid];
#pragma unroll 8
    for (int k = 0; k < D_PROJ; ++k)
        p = fmaf(hs[k], W2[(size_t)k * D_PROJ + tid], p);

    const float pn = block_reduce_128(p * p, red, tid);
    const float z  = p / fmaxf(sqrtf(pn), 1e-6f);
    Zh[(size_t)row * D_PROJ + tid] = (_Float16)z;
}

// -----------------------------------------------------------------------------
// Kernel 2: fused  sim = Zh @ Zh^T / T  with streaming logsumexp.
// One block (8 wave32) owns 16 rows; waves sweep disjoint 16-column tiles
// across all 8192 columns (64 iterations). K = 128 -> 4x
// v_wmma_f32_16x16x32_f16 per 16x16 tile. Rows of Z are unit-norm, so
// max_j sim[i,j] = 1/T and logsumexp is stable with the FIXED shift 1/T:
//   S_i = sum_j exp((dot_ij - 1)/T)
//   loss_i = (1 - dot_ii)/T + log(S_i) + log 2
// dot_ii is computed scalar in the preamble (pure f16-quantization correction)
// so the inner loop carries zero control flow.
// -----------------------------------------------------------------------------
__global__ __launch_bounds__(256) void simclr_lse_kernel(
    const _Float16* __restrict__ Zh, float* __restrict__ rowLoss)
{
    __shared__ float rowSum[16];
    __shared__ float diagv[16];

    const int tid  = threadIdx.x;
    const int wave = tid >> 5;
    const int lane = tid & 31;
    const int m    = lane & 15;   // row (A) / column (B) within 16-tile
    const int hi   = lane >> 4;   // half-wave selector
    const int i0   = blockIdx.x * 16;

    if (tid < 16) {
        rowSum[tid] = 0.f;
        // dot_ii = sum_k zh[i,k]^2 in f32 (diagonal of sim before 1/T scale)
        const _Float16* zr = Zh + (size_t)(i0 + tid) * D_PROJ;
        float d = 0.f;
#pragma unroll 8
        for (int k = 0; k < D_PROJ; ++k) {
            const float v = (float)zr[k];
            d = fmaf(v, v, d);
        }
        diagv[tid] = d;
    }
    __syncthreads();

    // Preload A fragments: rows i0..i0+15, K=128 in 4 chunks of 32.
    // A k order used here (matches B): lanes 0-15 hold K c*32+0..15,
    //                  lanes 16-31 hold K c*32+16..31.
    v16h aFrag[4];
    const _Float16* arow = Zh + (size_t)(i0 + m) * D_PROJ;
#pragma unroll
    for (int kk = 0; kk < 4; ++kk) {
        const _Float16* base = arow + kk * 32;
        const v8h lo  = *(const v8h*)(base + hi * 16);
        const v8h hic = *(const v8h*)(base + hi * 16 + 8);
#pragma unroll
        for (int t = 0; t < 8; ++t) { aFrag[kk][t] = lo[t]; aFrag[kk][t + 8] = hic[t]; }
    }

    float accS[8];
#pragma unroll
    for (int r = 0; r < 8; ++r) accS[r] = 0.f;

    for (int jb = 0; jb < N_ROWS; jb += 128) {
        const int j0 = jb + wave * 16;
        // B k order used here (matches A): lane = column, lanes 0-15 hold K 0..15,
        // lanes 16-31 hold K 16..31 of each 32-chunk -> one contiguous 32B load.
        const _Float16* brow = Zh + (size_t)(j0 + m) * D_PROJ;

        v8f c = {};
#pragma unroll
        for (int kk = 0; kk < 4; ++kk) {
            const v16h b = *(const v16h*)(brow + kk * 32 + hi * 16);
            c = __builtin_amdgcn_wmma_f32_16x16x32_f16(
                    /*neg_a=*/false, aFrag[kk], /*neg_b=*/false, b,
                    /*c_mod=*/(short)0, c, /*reuse_a=*/false, /*reuse_b=*/false);
        }

#pragma unroll
        for (int r = 0; r < 8; ++r)                      // tile elem (r+8*hi, m)
            accS[r] += __expf(fmaf(c[r], INV_TEMP, -INV_TEMP));
    }

#pragma unroll
    for (int r = 0; r < 8; ++r)
        atomicAdd(&rowSum[r + 8 * hi], accS[r]);         // ds_add_f32
    __syncthreads();

    if (tid < 16) {
        const float S = rowSum[tid];
        rowLoss[i0 + tid] =
            (1.0f - diagv[tid]) * INV_TEMP + __logf(S) + LOG2F_CONST;
    }
}

// -----------------------------------------------------------------------------
// Kernel 3: loss = mean(rowLoss)   (log 2 already folded per-row)
// -----------------------------------------------------------------------------
__global__ __launch_bounds__(256) void reduce_mean_kernel(
    const float* __restrict__ rowLoss, float* __restrict__ out)
{
    __shared__ float red[256];
    const int tid = threadIdx.x;
    float s = 0.f;
    for (int i = tid; i < N_ROWS; i += 256) s += rowLoss[i];
    red[tid] = s;
    __syncthreads();
#pragma unroll
    for (int st = 128; st > 0; st >>= 1) {
        if (tid < st) red[tid] += red[tid + st];
        __syncthreads();
    }
    if (tid == 0) out[0] = red[0] / (float)N_ROWS;
}

extern "C" void kernel_launch(void* const* d_in, const int* in_sizes, int n_in,
                              void* d_out, int out_size, void* d_ws, size_t ws_size,
                              hipStream_t stream) {
    const float* features = (const float*)d_in[0];
    const float* noise1   = (const float*)d_in[1];
    const float* noise2   = (const float*)d_in[2];
    const float* W1       = (const float*)d_in[3];
    const float* b1       = (const float*)d_in[4];
    const float* W2       = (const float*)d_in[5];
    const float* b2       = (const float*)d_in[6];

    _Float16* Zh   = (_Float16*)d_ws;                                // 2 MB
    float* rowLoss = (float*)((char*)d_ws +
                              (size_t)N_ROWS * D_PROJ * sizeof(_Float16));

    augment_project_kernel<<<N_ROWS, 128, 0, stream>>>(
        features, noise1, noise2, W1, b1, W2, b2, Zh);
    simclr_lse_kernel<<<N_ROWS / 16, 256, 0, stream>>>(Zh, rowLoss);
    reduce_mean_kernel<<<1, 256, 0, stream>>>(rowLoss, (float*)d_out);
}
